// H_gcn_43147241455751
// MI455X (gfx1250) — compile-verified, matrix operand order checked by reading, not run on hardware
//
#include <hip/hip_runtime.h>
#include <hip/hip_bf16.h>
#include <cmath>

typedef __attribute__((ext_vector_type(16))) _Float16 v16h;
typedef __attribute__((ext_vector_type(8)))  float    v8f;

#define D_HID 256           // hidden size (W1 cols, W2 rows)
#define D_IN  (2 * D_HID)   // 512 input features

// ---------------- init: zero aggregation buffers, deg=1 (self loop), accum=0 --
__global__ void k_init(float* __restrict__ deg, float* __restrict__ Hagg,
                       float* __restrict__ h2agg, float* __restrict__ accum,
                       int N) {
  long long idx    = (long long)blockIdx.x * blockDim.x + threadIdx.x;
  long long stride = (long long)gridDim.x * blockDim.x;
  long long total  = (long long)N * D_HID;
  for (long long i = idx; i < total; i += stride) {
    Hagg[i] = 0.0f;
    if (i < N) { deg[i] = 1.0f; h2agg[i] = 0.0f; }
    if (i == 0) accum[0] = 0.0f;
  }
}

// ---------------- degree: deg[col] += 1 per edge ------------------------------
__global__ void k_deg(const int* __restrict__ col, int E, float* __restrict__ deg) {
  int e = blockIdx.x * blockDim.x + threadIdx.x;
  if (e < E) atomicAdd(&deg[col[e]], 1.0f);
}

__global__ void k_dinv(const float* __restrict__ deg, float* __restrict__ dinv, int N) {
  int i = blockIdx.x * blockDim.x + threadIdx.x;
  if (i < N) dinv[i] = rsqrtf(deg[i]);   // deg >= 1 always (self loops)
}

// ---------------- W1 f32 -> f16, pre-swizzled into WMMA B-fragment layout -----
// W1s flat index o = ((tn*16 + kb)*32 + lane)*16 + i, where the (lane,i) -> (K,col)
// mapping matches the ISA 16-bit B-matrix 32x16 fragment layout:
//   K   = kb*32 + ((i>>3)&1)*16 + ((lane>>4)&1)*8 + (i&7)
//   col = tn*16 + (lane&15)
__global__ void k_w1s(const float* __restrict__ W1, _Float16* __restrict__ W1s, int n) {
  int o = blockIdx.x * blockDim.x + threadIdx.x;
  if (o >= n) return;
  int i    = o & 15;
  int lane = (o >> 4) & 31;
  int kb   = (o >> 9) & 15;
  int tn   = o >> 13;
  int K    = kb * 32 + ((i >> 3) & 1) * 16 + ((lane >> 4) & 1) * 8 + (i & 7);
  int col  = tn * 16 + (lane & 15);
  W1s[o] = (_Float16)W1[K * D_HID + col];
}

// ---------------- GEMM: H1 = x @ W1 via v_wmma_f32_16x16x32_f16 ---------------
// One block per 16-row strip of x; 8 waves each own two 16-col strips
// (2 accumulators), so each A fragment feeds 2 WMMAs and x is read once.
__global__ void __launch_bounds__(256) k_gemm(const float* __restrict__ x,
                                              const _Float16* __restrict__ W1s,
                                              float* __restrict__ H1, int N) {
  int wave = threadIdx.x >> 5;
  int lane = threadIdx.x & 31;
  int m0 = blockIdx.x << 4;
  int lanelo = lane & 15;
  int lanehi = (lane >> 4) & 1;

  int rowA = m0 + lanelo; if (rowA >= N) rowA = N - 1;   // clamped load row
  const float* xrow = x + (long long)rowA * D_IN;

  int tn0 = wave * 2, tn1 = wave * 2 + 1;                // two col strips / wave
  const v16h* Bf = reinterpret_cast<const v16h*>(W1s);

  v8f c0 = {}, c1 = {};
  #pragma unroll 4
  for (int kb = 0; kb < 16; ++kb) {
    int kh = kb * 32 + lanehi * 8;
    // A fragment: two contiguous 8-float chunks (K = kh..kh+7, kh+16..kh+23)
    const float4* pa = reinterpret_cast<const float4*>(xrow + kh);
    float4 a0 = pa[0], a1 = pa[1], a2 = pa[4], a3 = pa[5];
    v16h a;
    a[0]=(_Float16)a0.x;  a[1]=(_Float16)a0.y;  a[2]=(_Float16)a0.z;  a[3]=(_Float16)a0.w;
    a[4]=(_Float16)a1.x;  a[5]=(_Float16)a1.y;  a[6]=(_Float16)a1.z;  a[7]=(_Float16)a1.w;
    a[8]=(_Float16)a2.x;  a[9]=(_Float16)a2.y;  a[10]=(_Float16)a2.z; a[11]=(_Float16)a2.w;
    a[12]=(_Float16)a3.x; a[13]=(_Float16)a3.y; a[14]=(_Float16)a3.z; a[15]=(_Float16)a3.w;
    // B fragments: contiguous 32B per lane from the pre-swizzled W1s
    v16h b0 = Bf[(tn0 * 16 + kb) * 32 + lane];
    v16h b1 = Bf[(tn1 * 16 + kb) * 32 + lane];
    c0 = __builtin_amdgcn_wmma_f32_16x16x32_f16(false, a, false, b0,
                                                (short)0, c0, false, false);
    c1 = __builtin_amdgcn_wmma_f32_16x16x32_f16(false, a, false, b1,
                                                (short)0, c1, false, false);
  }
  #pragma unroll
  for (int r = 0; r < 8; ++r) {
    int row = m0 + r + lanehi * 8;
    if (row < N) {
      H1[(long long)row * D_HID + tn0 * 16 + lanelo] = c0[r];
      H1[(long long)row * D_HID + tn1 * 16 + lanelo] = c1[r];
    }
  }
}

// ---------------- aggregation 1: Hagg[c] += dinv[r]*dinv[c] * H1[r]  ----------
// One wave per edge (t >= E means self-loop node t-E); 8 channels per lane.
__global__ void __launch_bounds__(256) k_agg1(const int* __restrict__ ei, int E, int N,
                                              const float* __restrict__ dinv,
                                              const float* __restrict__ H1,
                                              float* __restrict__ Hagg) {
  int t = blockIdx.x * 8 + (threadIdx.x >> 5);
  int lane = threadIdx.x & 31;
  if (t >= E + N) return;
  int r, c;
  if (t < E) { r = ei[t]; c = ei[E + t]; } else { r = t - E; c = r; }
  float w = dinv[r] * dinv[c];
  const float4* src = reinterpret_cast<const float4*>(H1 + (long long)r * D_HID + lane * 8);
  float4 v0 = src[0], v1 = src[1];
  float* dst = Hagg + (long long)c * D_HID + lane * 8;
  atomicAdd(dst + 0, w * v0.x); atomicAdd(dst + 1, w * v0.y);
  atomicAdd(dst + 2, w * v0.z); atomicAdd(dst + 3, w * v0.w);
  atomicAdd(dst + 4, w * v1.x); atomicAdd(dst + 5, w * v1.y);
  atomicAdd(dst + 6, w * v1.z); atomicAdd(dst + 7, w * v1.w);
}

// ---------------- layer 2: h2[i] = tanh(Hagg[i,:]+b1) . W2  (wave per node) ---
__global__ void __launch_bounds__(256) k_layer2(const float* __restrict__ Hagg,
                                                const float* __restrict__ b1,
                                                const float* __restrict__ W2,
                                                float* __restrict__ h2, int N) {
  int i = blockIdx.x * 8 + (threadIdx.x >> 5);
  int lane = threadIdx.x & 31;
  if (i >= N) return;
  const float* hrow = Hagg + (long long)i * D_HID;
  float s = 0.0f;
  #pragma unroll
  for (int j = 0; j < 8; ++j) {
    int ch = lane * 8 + j;
    s += tanhf(hrow[ch] + b1[ch]) * W2[ch];
  }
  #pragma unroll
  for (int off = 16; off > 0; off >>= 1) s += __shfl_xor(s, off, 32);
  if (lane == 0) h2[i] = s;
}

// ---------------- aggregation 2 (scalar per node) -----------------------------
__global__ void k_agg2(const int* __restrict__ ei, int E, int N,
                       const float* __restrict__ dinv, const float* __restrict__ h2,
                       float* __restrict__ h2agg) {
  int t = blockIdx.x * blockDim.x + threadIdx.x;
  if (t >= E + N) return;
  int r, c;
  if (t < E) { r = ei[t]; c = ei[E + t]; } else { r = t - E; c = r; }
  atomicAdd(&h2agg[c], dinv[r] * dinv[c] * h2[r]);
}

// ---------------- norm: accum += sum (h2agg[i]+b2)^2 --------------------------
__global__ void k_norm(const float* __restrict__ h2agg, const float* __restrict__ b2,
                       float* __restrict__ accum, int N) {
  int idx = blockIdx.x * blockDim.x + threadIdx.x;
  int stride = gridDim.x * blockDim.x;
  float bb = b2[0];
  float s = 0.0f;
  for (int i = idx; i < N; i += stride) {
    float v = h2agg[i] + bb;
    s += v * v;
  }
  #pragma unroll
  for (int off = 16; off > 0; off >>= 1) s += __shfl_xor(s, off, 32);
  if ((threadIdx.x & 31) == 0) atomicAdd(accum, s);
}

__global__ void k_sqrt(const float* __restrict__ accum, float* __restrict__ out) {
  out[0] = sqrtf(accum[0]);
}

// -----------------------------------------------------------------------------
extern "C" void kernel_launch(void* const* d_in, const int* in_sizes, int n_in,
                              void* d_out, int out_size, void* d_ws, size_t ws_size,
                              hipStream_t stream) {
  const float* x  = (const float*)d_in[0];
  const int*   ei = (const int*)d_in[1];
  const float* W1 = (const float*)d_in[2];
  const float* b1 = (const float*)d_in[3];
  const float* W2 = (const float*)d_in[4];
  const float* b2 = (const float*)d_in[5];
  float* out = (float*)d_out;

  int N = in_sizes[0] / D_IN;   // 100000
  int E = in_sizes[1] / 2;      // 3200000

  char* ws = (char*)d_ws;
  size_t o = 0;
  _Float16* W1s  = (_Float16*)(ws + o); o += (size_t)D_IN * D_HID * sizeof(_Float16);
  float* H1      = (float*)(ws + o);    o += (size_t)N * D_HID * sizeof(float);
  float* Hagg    = (float*)(ws + o);    o += (size_t)N * D_HID * sizeof(float);
  float* deg     = (float*)(ws + o);    o += (size_t)N * sizeof(float);
  float* dinv    = (float*)(ws + o);    o += (size_t)N * sizeof(float);
  float* h2      = (float*)(ws + o);    o += (size_t)N * sizeof(float);
  float* h2agg   = (float*)(ws + o);    o += (size_t)N * sizeof(float);
  float* accum   = (float*)(ws + o);    o += 256;

  k_init<<<4096, 256, 0, stream>>>(deg, Hagg, h2agg, accum, N);
  k_deg<<<(E + 255) / 256, 256, 0, stream>>>(ei + E, E, deg);
  k_dinv<<<(N + 255) / 256, 256, 0, stream>>>(deg, dinv, N);
  k_w1s<<<(D_IN * D_HID + 255) / 256, 256, 0, stream>>>(W1, W1s, D_IN * D_HID);

  int mtiles = (N + 15) / 16;
  k_gemm<<<mtiles, 256, 0, stream>>>(x, W1s, H1, N);

  k_agg1<<<(E + N + 7) / 8, 256, 0, stream>>>(ei, E, N, dinv, H1, Hagg);
  k_layer2<<<(N + 7) / 8, 256, 0, stream>>>(Hagg, b1, W2, h2, N);
  k_agg2<<<(E + N + 255) / 256, 256, 0, stream>>>(ei, E, N, dinv, h2, h2agg);
  k_norm<<<1024, 256, 0, stream>>>(h2agg, b2, accum, N);
  k_sqrt<<<1, 1, 0, stream>>>(accum, out);
}
